// Gemma3Attention_47339129536774
// MI455X (gfx1250) — compile-verified, matrix operand order checked by reading, not run on hardware
//
#include <hip/hip_runtime.h>
#include <hip/hip_bf16.h>
#include <stdint.h>

// ---------------- problem constants ----------------
#define B_   4
#define S_   2048
#define E_   1152
#define H_   8
#define KV_  2
#define D_   128
#define G_   (H_ / KV_)
#define SW_  1024
#define NQKV 1536            // H*D + 2*KV*D
#define HD_  (H_ * D_)       // 1024
#define SPAD (S_ + 16)       // padded key length
#define EPS_ 1e-6f
#define ROPE_BASE_ 10000.0f
#define QSCALE 0.08838834764831845f   // 1/sqrt(128)
#define NEG_BIG (-1e30f)

typedef __attribute__((ext_vector_type(16))) __bf16 v16bf;
typedef __attribute__((ext_vector_type(8)))  __bf16 v8bf;
typedef __attribute__((ext_vector_type(4)))  __bf16 v4bf;
typedef __attribute__((ext_vector_type(8)))  float  v8f;

static __device__ inline v8f vzero8() {
  v8f r;
#pragma unroll
  for (int i = 0; i < 8; ++i) r[i] = 0.0f;
  return r;
}

static __device__ inline v8f wmma_bf16(v16bf a, v16bf b, v8f c) {
  // D = A(16x32 bf16) * B(32x16 bf16) + C(16x16 f32)
  return __builtin_amdgcn_wmma_f32_16x16x32_bf16(
      /*neg_a=*/false, a, /*neg_b=*/false, b,
      /*c_mod=*/(short)0, c, /*reuse_a=*/false, /*reuse_b=*/false);
}

// Fragment loader for 16-bit A (16x32) / mirrored B (32x16):
// lane (row/col) = lane&15; half = lane>>4.
// Element e<8  -> k = half*8 + e ; e>=8 -> k = 16 + half*8 + (e-8)
static __device__ inline v16bf frag_ld(const __bf16* rowbase, int half) {
  const v8bf* p = (const v8bf*)rowbase;
  v8bf lo = p[half];
  v8bf hi = p[2 + half];
  v16bf r;
#pragma unroll
  for (int i = 0; i < 8; ++i) { r[i] = lo[i]; r[8 + i] = hi[i]; }
  return r;
}

// reductions over the 16 lanes sharing one C-layout half (masks stay in-half)
static __device__ inline float half_red_max(float v) {
#pragma unroll
  for (int m = 1; m < 16; m <<= 1) v = fmaxf(v, __shfl_xor(v, m, 32));
  return v;
}
static __device__ inline float half_red_sum(float v) {
#pragma unroll
  for (int m = 1; m < 16; m <<= 1) v += __shfl_xor(v, m, 32);
  return v;
}
static __device__ inline float wave_red_sum(float v) {
#pragma unroll
  for (int m = 1; m < 32; m <<= 1) v += __shfl_xor(v, m, 32);
  return v;
}

// CDNA5 async global->LDS copy of one 16-byte chunk (ASYNCcnt-tracked).
static __device__ inline void async_ld_b128(uint32_t lds_off, const void* gptr) {
  uint64_t ga = (uint64_t)(uintptr_t)gptr;
  asm volatile("global_load_async_to_lds_b128 %0, %1, off"
               :: "v"(lds_off), "v"(ga) : "memory");
}
static __device__ inline void wait_async0() {
  asm volatile("s_wait_asynccnt 0" ::: "memory");
}

// ---------------- kernel 1: weight convert/transpose to bf16 ----------------
__global__ void __launch_bounds__(256) convert_kernel(
    const float* __restrict__ Wq, const float* __restrict__ Wk,
    const float* __restrict__ Wv, const float* __restrict__ Wo,
    __bf16* __restrict__ WqkvT, __bf16* __restrict__ WoT) {
  const int NA = NQKV * E_;
  const int NB = E_ * HD_;
  int idx = blockIdx.x * 256 + threadIdx.x;
  if (idx < NA) {
    int n = idx / E_;
    int k = idx - n * E_;
    float v;
    if (n < HD_)              v = Wq[(size_t)k * HD_ + n];
    else if (n < HD_ + 256)   v = Wk[(size_t)k * (KV_ * D_) + (n - HD_)];
    else                      v = Wv[(size_t)k * (KV_ * D_) + (n - HD_ - 256)];
    WqkvT[idx] = (__bf16)v;
  } else if (idx < NA + NB) {
    int j = idx - NA;
    int n = j / HD_;
    int k = j - n * HD_;
    WoT[j] = (__bf16)Wo[(size_t)k * E_ + n];
  }
}

// ---------------- kernel 1b: hidden f32 -> bf16 (one-time) ----------------
__global__ void __launch_bounds__(256) cvt_hidden_kernel(
    const float* __restrict__ src, __bf16* __restrict__ dst) {
  int idx = blockIdx.x * 256 + threadIdx.x;   // one float4 per thread
  float4 x = ((const float4*)src)[idx];
  v4bf o;
  o[0] = (__bf16)x.x; o[1] = (__bf16)x.y; o[2] = (__bf16)x.z; o[3] = (__bf16)x.w;
  ((v4bf*)dst)[idx] = o;
}

// ---------------- kernel 2: QKV projection GEMM ---------------------------
// 16x64 tile per wave. K-loop unrolled x2 with structural double buffering:
// no rotating-register copies, loads of chunk i+1 overlap WMMAs of chunk i.
__global__ void __launch_bounds__(256) qkv_gemm_kernel(
    const __bf16* __restrict__ hid, const __bf16* __restrict__ WT,
    float* __restrict__ out) {
  int tid = threadIdx.x;
  int wl = tid >> 5, lane = tid & 31, half = lane >> 4, r = lane & 15;
  int wave = blockIdx.x * 8 + wl;
  const int NG = NQKV / 64;          // 24 n-groups of 64
  int ng = wave % NG, mt = wave / NG;
  int m0 = mt * 16, n0 = ng * 64;
  const __bf16* arow = hid + (size_t)(m0 + r) * E_;
  const __bf16* brow0 = WT + (size_t)(n0 + r) * E_;
  v8f c[4];
#pragma unroll
  for (int j = 0; j < 4; ++j) c[j] = vzero8();

  v16bf a0, a1, b0[4], b1[4];
  a0 = frag_ld(arow, half);
#pragma unroll
  for (int j = 0; j < 4; ++j) b0[j] = frag_ld(brow0 + (size_t)j * 16 * E_, half);

  const int NPAIR = (E_ / 32) / 2;   // 18
#pragma unroll 1
  for (int it = 0; it < NPAIR - 1; ++it) {
    int kA = it * 64 + 32, kB = it * 64 + 64;
    a1 = frag_ld(arow + kA, half);
#pragma unroll
    for (int j = 0; j < 4; ++j) b1[j] = frag_ld(brow0 + (size_t)j * 16 * E_ + kA, half);
#pragma unroll
    for (int j = 0; j < 4; ++j) c[j] = wmma_bf16(a0, b0[j], c[j]);
    __builtin_prefetch(arow + kB + 256, 0, 1);
    a0 = frag_ld(arow + kB, half);
#pragma unroll
    for (int j = 0; j < 4; ++j) b0[j] = frag_ld(brow0 + (size_t)j * 16 * E_ + kB, half);
#pragma unroll
    for (int j = 0; j < 4; ++j) c[j] = wmma_bf16(a1, b1[j], c[j]);
  }
  // tail: buf0 holds chunk at E_-64; load final chunk at E_-32
  a1 = frag_ld(arow + E_ - 32, half);
#pragma unroll
  for (int j = 0; j < 4; ++j) b1[j] = frag_ld(brow0 + (size_t)j * 16 * E_ + E_ - 32, half);
#pragma unroll
  for (int j = 0; j < 4; ++j) c[j] = wmma_bf16(a0, b0[j], c[j]);
#pragma unroll
  for (int j = 0; j < 4; ++j) c[j] = wmma_bf16(a1, b1[j], c[j]);

#pragma unroll
  for (int j = 0; j < 4; ++j)
#pragma unroll
    for (int v = 0; v < 8; ++v) {
      int m = v + half * 8;
      out[(size_t)(m0 + m) * NQKV + n0 + j * 16 + r] = c[j][v];
    }
}

// ---------------- kernel 3: RMSNorm + RoPE + layout ----------------
__global__ void __launch_bounds__(256) normrope_kernel(
    const float* __restrict__ raw, const float* __restrict__ qw,
    const float* __restrict__ kw, __bf16* __restrict__ Q,
    __bf16* __restrict__ Kb, __bf16* __restrict__ Vt) {
  int tid = threadIdx.x;
  int wl = tid >> 5, lane = tid & 31;
  int token = blockIdx.x * 8 + wl;
  int b = token / S_, sIdx = token - b * S_;
  const float* row = raw + (size_t)token * NQKV;
  int d0 = lane * 4;
  bool firstHalf = (d0 < (D_ / 2));

  float cosv[4], sinv[4], qwv[4], kwv[4];
#pragma unroll
  for (int i = 0; i < 4; ++i) {
    int d = d0 + i;
    int fi = d & 63;
    float ang = (float)sIdx * __powf(ROPE_BASE_, -((float)(2 * fi)) / (float)D_);
    __sincosf(ang, &sinv[i], &cosv[i]);
    qwv[i] = qw[d];
    kwv[i] = kw[d];
  }

#pragma unroll
  for (int h = 0; h < H_; ++h) {
    float4 x4 = *(const float4*)(row + h * D_ + d0);
    float x[4] = {x4.x, x4.y, x4.z, x4.w};
    float ss = x[0]*x[0] + x[1]*x[1] + x[2]*x[2] + x[3]*x[3];
    ss = wave_red_sum(ss);
    float rms = rsqrtf(ss * (1.0f / (float)D_) + EPS_);
    __bf16* qdst = Q + ((size_t)(b * H_ + h) * S_ + sIdx) * D_ + d0;
#pragma unroll
    for (int i = 0; i < 4; ++i) {
      float xn = x[i] * rms * qwv[i];
      float pr = __shfl_xor(xn, 16, 32);
      float rot = firstHalf ? -pr : pr;
      qdst[i] = (__bf16)((xn * cosv[i] + rot * sinv[i]) * QSCALE);
    }
  }
#pragma unroll
  for (int kv = 0; kv < KV_; ++kv) {
    float4 x4 = *(const float4*)(row + HD_ + kv * D_ + d0);
    float x[4] = {x4.x, x4.y, x4.z, x4.w};
    float ss = x[0]*x[0] + x[1]*x[1] + x[2]*x[2] + x[3]*x[3];
    ss = wave_red_sum(ss);
    float rms = rsqrtf(ss * (1.0f / (float)D_) + EPS_);
    __bf16* kdst = Kb + ((size_t)(b * KV_ + kv) * SPAD + sIdx) * D_ + d0;
#pragma unroll
    for (int i = 0; i < 4; ++i) {
      float xn = x[i] * rms * kwv[i];
      float pr = __shfl_xor(xn, 16, 32);
      float rot = firstHalf ? -pr : pr;
      kdst[i] = (__bf16)(xn * cosv[i] + rot * sinv[i]);
    }
  }
#pragma unroll
  for (int kv = 0; kv < KV_; ++kv) {
    float4 x4 = *(const float4*)(row + HD_ + KV_ * D_ + kv * D_ + d0);
    float x[4] = {x4.x, x4.y, x4.z, x4.w};
#pragma unroll
    for (int i = 0; i < 4; ++i)
      Vt[((size_t)(b * KV_ + kv) * D_ + d0 + i) * SPAD + sIdx] = (__bf16)x[i];
  }
}

// ---------------- kernel 4: zero padding region of K / Vt ----------------
__global__ void __launch_bounds__(256) zeropad_kernel(__bf16* __restrict__ Kb,
                                                      __bf16* __restrict__ Vt) {
  int idx = blockIdx.x * 256 + threadIdx.x;   // B*KV*16*D = 16384
  int d = idx & (D_ - 1);
  int p = (idx >> 7) & 15;
  int bk = idx >> 11;
  Kb[((size_t)bk * SPAD + S_ + p) * D_ + d] = (__bf16)0.0f;
  Vt[((size_t)bk * D_ + d) * SPAD + S_ + p] = (__bf16)0.0f;
}

// ---------------- kernel 5: sliding-window flash attention (WMMA) -------
__global__ void __launch_bounds__(256) attn_kernel(
    const __bf16* __restrict__ Q, const __bf16* __restrict__ Kb,
    const __bf16* __restrict__ Vt, __bf16* __restrict__ AO) {
  __shared__ __align__(16) __bf16 Qlds[8][16 * D_];   // 32 KB
  __shared__ __align__(16) __bf16 Plds[8][16 * 32];   //  8 KB
  int tid = threadIdx.x;
  int wl = tid >> 5, lane = tid & 31, half = lane >> 4, r = lane & 15;
  int wave = blockIdx.x * 8 + wl;
  int qt = wave & 127;            // S/16 = 128
  int bh = wave >> 7;             // 0..31
  int b = bh >> 3, h = bh & 7;
  int kvh = h >> 2;               // G = 4
  int i0 = qt * 16;

  const __bf16* Qp = Q + ((size_t)(b * H_ + h) * S_ + i0) * D_;
  const __bf16* Kp = Kb + (size_t)(b * KV_ + kvh) * SPAD * D_;
  const __bf16* Vp = Vt + (size_t)(b * KV_ + kvh) * D_ * SPAD;

  // ---- stage contiguous 4KB Q tile into LDS via async-to-LDS DMA ----
  __bf16* ql = Qlds[wl];
  {
    uint32_t base = (uint32_t)(uintptr_t)ql;      // LDS byte offset (low 32b)
#pragma unroll
    for (int u = 0; u < 8; ++u) {
      int chunk = lane + u * 32;                  // 256 x 16B = 4KB
      async_ld_b128(base + chunk * 16, (const char*)Qp + chunk * 16);
    }
    wait_async0();
  }
  v16bf aq[4];
#pragma unroll
  for (int c = 0; c < 4; ++c) aq[c] = frag_ld(ql + r * D_ + c * 32, half);

  float mrow[8], lrow[8];
  v8f acc[8];
#pragma unroll
  for (int v = 0; v < 8; ++v) { mrow[v] = NEG_BIG; lrow[v] = 0.0f; }
#pragma unroll
  for (int dc = 0; dc < 8; ++dc) acc[dc] = vzero8();

  int ktlo = (i0 >= SW_) ? ((i0 - SW_) >> 4) : 0;
  for (int kt = ktlo; kt <= qt; kt += 2) {
    // ---- issue ALL loads for this 32-key block up front (K then V) ----
    v16bf bk[2][4];
#pragma unroll
    for (int t = 0; t < 2; ++t) {
      const __bf16* krow = Kp + (size_t)((kt + t) * 16 + r) * D_;
#pragma unroll
      for (int c = 0; c < 4; ++c) bk[t][c] = frag_ld(krow + c * 32, half);
    }
    v16bf bv[8];
#pragma unroll
    for (int dc = 0; dc < 8; ++dc)
      bv[dc] = frag_ld(Vp + (size_t)(dc * 16 + r) * SPAD + kt * 16, half);

    // ---- scores (V loads still in flight; softmax VALU hides them) ----
    v8f s[2];
#pragma unroll
    for (int t = 0; t < 2; ++t) {
      v8f cacc = vzero8();
#pragma unroll
      for (int c = 0; c < 4; ++c) cacc = wmma_bf16(aq[c], bk[t][c], cacc);
      s[t] = cacc;
    }
    float p0v[8], p1v[8], scl[8];
#pragma unroll
    for (int v = 0; v < 8; ++v) {
      int m = v + half * 8;
      int i = i0 + m;
#pragma unroll
      for (int t = 0; t < 2; ++t) {
        int j = (kt + t) * 16 + r;
        float sv = s[t][v];
        if (j > i || j < i - SW_) sv = NEG_BIG;
        s[t][v] = sv;
      }
      float rmax = half_red_max(fmaxf(s[0][v], s[1][v]));
      float mnew = fmaxf(mrow[v], rmax);
      float sc = __expf(mrow[v] - mnew);
      mrow[v] = mnew;
      scl[v] = sc;
      float p0 = __expf(s[0][v] - mnew);
      float p1 = __expf(s[1][v] - mnew);
      p0v[v] = p0; p1v[v] = p1;
      lrow[v] = lrow[v] * sc + half_red_sum(p0 + p1);
    }
#pragma unroll
    for (int dc = 0; dc < 8; ++dc)
#pragma unroll
      for (int v = 0; v < 8; ++v) acc[dc][v] *= scl[v];

    // ---- P: C-layout -> LDS -> A fragment ----
    __bf16* pl = Plds[wl];
#pragma unroll
    for (int v = 0; v < 8; ++v) {
      int m = v + half * 8;
      pl[m * 32 + r]      = (__bf16)p0v[v];
      pl[m * 32 + 16 + r] = (__bf16)p1v[v];
    }
    v16bf pa = frag_ld(pl + r * 32, half);

#pragma unroll
    for (int dc = 0; dc < 8; ++dc) acc[dc] = wmma_bf16(pa, bv[dc], acc[dc]);
  }

  float inv[8];
#pragma unroll
  for (int v = 0; v < 8; ++v) inv[v] = 1.0f / lrow[v];
  size_t row0 = (size_t)b * S_ + i0;
#pragma unroll
  for (int dc = 0; dc < 8; ++dc)
#pragma unroll
    for (int v = 0; v < 8; ++v) {
      int m = v + half * 8;
      AO[(row0 + m) * HD_ + h * D_ + dc * 16 + r] = (__bf16)(acc[dc][v] * inv[v]);
    }
}

// ---------------- kernel 6: output projection GEMM (pipelined 16x64) -----
__global__ void __launch_bounds__(256) proj_gemm_kernel(
    const __bf16* __restrict__ AO, const __bf16* __restrict__ WoT,
    float* __restrict__ out) {
  int tid = threadIdx.x;
  int wl = tid >> 5, lane = tid & 31, half = lane >> 4, r = lane & 15;
  int wave = blockIdx.x * 8 + wl;
  const int NG = E_ / 64;            // 18 n-groups of 64
  int ng = wave % NG, mt = wave / NG;
  int m0 = mt * 16, n0 = ng * 64;
  const __bf16* arow = AO + (size_t)(m0 + r) * HD_;
  const __bf16* brow0 = WoT + (size_t)(n0 + r) * HD_;
  v8f c[4];
#pragma unroll
  for (int j = 0; j < 4; ++j) c[j] = vzero8();

  v16bf a0, a1, b0[4], b1[4];
  a0 = frag_ld(arow, half);
#pragma unroll
  for (int j = 0; j < 4; ++j) b0[j] = frag_ld(brow0 + (size_t)j * 16 * HD_, half);

  const int NPAIR = (HD_ / 32) / 2;  // 16
#pragma unroll 1
  for (int it = 0; it < NPAIR - 1; ++it) {
    int kA = it * 64 + 32, kB = it * 64 + 64;
    a1 = frag_ld(arow + kA, half);
#pragma unroll
    for (int j = 0; j < 4; ++j) b1[j] = frag_ld(brow0 + (size_t)j * 16 * HD_ + kA, half);
#pragma unroll
    for (int j = 0; j < 4; ++j) c[j] = wmma_bf16(a0, b0[j], c[j]);
    __builtin_prefetch(arow + kB + 256, 0, 1);
    a0 = frag_ld(arow + kB, half);
#pragma unroll
    for (int j = 0; j < 4; ++j) b0[j] = frag_ld(brow0 + (size_t)j * 16 * HD_ + kB, half);
#pragma unroll
    for (int j = 0; j < 4; ++j) c[j] = wmma_bf16(a1, b1[j], c[j]);
  }
  a1 = frag_ld(arow + HD_ - 32, half);
#pragma unroll
  for (int j = 0; j < 4; ++j) b1[j] = frag_ld(brow0 + (size_t)j * 16 * HD_ + HD_ - 32, half);
#pragma unroll
  for (int j = 0; j < 4; ++j) c[j] = wmma_bf16(a0, b0[j], c[j]);
#pragma unroll
  for (int j = 0; j < 4; ++j) c[j] = wmma_bf16(a1, b1[j], c[j]);

#pragma unroll
  for (int j = 0; j < 4; ++j)
#pragma unroll
    for (int v = 0; v < 8; ++v) {
      int m = v + half * 8;
      out[(size_t)(m0 + m) * E_ + n0 + j * 16 + r] = c[j][v];
    }
}

// ---------------- host orchestration ----------------
extern "C" void kernel_launch(void* const* d_in, const int* in_sizes, int n_in,
                              void* d_out, int out_size, void* d_ws, size_t ws_size,
                              hipStream_t stream) {
  const float* hidden = (const float*)d_in[0];
  const float* Wq     = (const float*)d_in[1];
  const float* Wk     = (const float*)d_in[2];
  const float* Wv     = (const float*)d_in[3];
  const float* Wo     = (const float*)d_in[4];
  const float* qw     = (const float*)d_in[5];
  const float* kw     = (const float*)d_in[6];
  float* out = (float*)d_out;

  char* ws = (char*)d_ws;
  size_t off = 0;
  auto take = [&](size_t bytes) {
    size_t o = off;
    off += (bytes + 255) & ~(size_t)255;
    return o;
  };
  __bf16* WqkvT = (__bf16*)(ws + take((size_t)NQKV * E_ * 2));
  __bf16* WoT   = (__bf16*)(ws + take((size_t)E_ * HD_ * 2));
  __bf16* Hb    = (__bf16*)(ws + take((size_t)B_ * S_ * E_ * 2));
  float*  QKVr  = (float*) (ws + take((size_t)B_ * S_ * NQKV * 4));
  __bf16* Qb    = (__bf16*)(ws + take((size_t)B_ * H_ * S_ * D_ * 2));
  __bf16* Kb    = (__bf16*)(ws + take((size_t)B_ * KV_ * SPAD * D_ * 2));
  __bf16* Vt    = (__bf16*)(ws + take((size_t)B_ * KV_ * D_ * SPAD * 2));
  __bf16* AO    = (__bf16*)(ws + take((size_t)B_ * S_ * HD_ * 2));
  (void)ws_size; (void)in_sizes; (void)n_in; (void)out_size;

  const int ntok = B_ * S_;                    // 8192
  {
    int total = NQKV * E_ + E_ * HD_;
    convert_kernel<<<(total + 255) / 256, 256, 0, stream>>>(Wq, Wk, Wv, Wo, WqkvT, WoT);
  }
  cvt_hidden_kernel<<<(ntok * E_ / 4) / 256, 256, 0, stream>>>(hidden, Hb);
  // QKV GEMM: (8192/16) m-tiles * (1536/64) n-groups = 12288 waves
  qkv_gemm_kernel<<<(ntok / 16) * (NQKV / 64) / 8, 256, 0, stream>>>(Hb, WqkvT, QKVr);
  normrope_kernel<<<ntok / 8, 256, 0, stream>>>(QKVr, qw, kw, Qb, Kb, Vt);
  zeropad_kernel<<<(B_ * KV_ * 16 * D_) / 256, 256, 0, stream>>>(Kb, Vt);
  attn_kernel<<<(B_ * H_ * (S_ / 16)) / 8, 256, 0, stream>>>(Qb, Kb, Vt, AO);
  // out projection: (8192/16) * (1152/64) = 9216 waves
  proj_gemm_kernel<<<(ntok / 16) * (E_ / 64) / 8, 256, 0, stream>>>(AO, WoT, out);
}
